// ForexActorNeuralNetwork_18476949308083
// MI455X (gfx1250) — matrix operand order repaired and checked by reading, not compiled, not run on hardware
//
#include <hip/hip_runtime.h>
#include <hip/hip_bf16.h>
#include <math.h>

// ---------------------------------------------------------------------------
// Types for CDNA5 WMMA (wave32): V_WMMA_F32_16X16X32_BF16
// ---------------------------------------------------------------------------
typedef __attribute__((ext_vector_type(16))) __bf16 v16bf;
typedef __attribute__((ext_vector_type(8)))  __bf16 v8bf;
typedef __attribute__((ext_vector_type(8)))  float  v8f;

static __device__ __forceinline__ v8f wmma_bf16(v16bf a, v16bf b, v8f c) {
    // 8 args: (neg_a, A, neg_b, B, c_mod, C, reuse_a, reuse_b)
    return __builtin_amdgcn_wmma_f32_16x16x32_bf16(false, a, false, b, (short)0, c,
                                                   false, false);
}

// A-matrix fragment (16x32 bf16) from a row-major bf16 buffer.
// Per ISA 7.12.2: lanes 0-15 hold row=lane, K = {0..7, 16..23} (+kt*32);
// lanes 16-31 hold row=lane-16, K = {8..15, 24..31}. `p` points at
// row*ldK + kt*32 + kbase where kbase = (lane<16 ? 0 : 8). Two contiguous
// 8-halve (16B) runs at p and p+16.
static __device__ __forceinline__ v16bf load_afrag(const __bf16* p) {
    v8bf lo = *(const v8bf*)(p);
    v8bf hi = *(const v8bf*)(p + 16);
    return __builtin_shufflevector(lo, hi, 0, 1, 2, 3, 4, 5, 6, 7,
                                   8, 9, 10, 11, 12, 13, 14, 15);
}

// B-matrix fragment (32x16 bf16, column n per lane) from row-major f32
// weights W[n][k]: per lane two 8-value K runs at p and p+16, p at
// kt*32 + (lane<16 ? 0 : 8). Converted f32 -> bf16 on the fly.
static __device__ __forceinline__ v16bf load_bfrag_f32(const float* p) {
    v16bf r;
#pragma unroll
    for (int i = 0; i < 16; ++i) r[i] = (__bf16)p[i < 8 ? i : i + 8];
    return r;
}

static __device__ __forceinline__ float fast_sigmoid(float x) {
    return 1.0f / (1.0f + __expf(-x));
}
static __device__ __forceinline__ float fast_tanh(float x) {
    return 1.0f - 2.0f / (__expf(2.0f * x) + 1.0f);
}

// ---------------------------------------------------------------------------
// Kernel 1: gate pre-compute  xg[m][n] = sum_k in[m][k]*Wih[n][k] + bih[n]+bhh[n]
//   m = b*T + t flattened (M = 466944), n in [0,512), K = 32 or 128.
//   Block: 256 threads (8 waves). Each block: 128 M-rows (wave w -> rows
//   w*16..w*16+15); each wave loops all 32 N-tiles. Activations staged as
//   bf16 in LDS; Wih fragments streamed from L2 (Wih is tiny, stays hot).
//   xg is write-once/read-once and bigger than L2 -> non-temporal stores.
// ---------------------------------------------------------------------------
template <int K>
__global__ void lstm_gates_kernel(const float* __restrict__ in,
                                  const float* __restrict__ Wih,
                                  const float* __restrict__ bih,
                                  const float* __restrict__ bhh,
                                  float* __restrict__ xg) {
    __shared__ __attribute__((aligned(16))) __bf16 sX[128 * K];
    __shared__ float sBias[512];

    const int tid  = threadIdx.x;
    const int lane = tid & 31;
    const int w    = tid >> 5;
    const size_t m0 = (size_t)blockIdx.x * 128;

    for (int i = tid; i < 512; i += 256) sBias[i] = bih[i] + bhh[i];
    for (int i = tid; i < 128 * K; i += 256) sX[i] = (__bf16)in[m0 * K + i];
    __syncthreads();

    const int l15    = lane & 15;
    const int kbaseA = (lane < 16) ? 0 : 8;
    const int kbaseB = (lane < 16) ? 0 : 8;
    const int mel    = (lane & 16) ? 8 : 0;
    const int rowA   = w * 16 + l15;

    constexpr int NKT = K / 32;
    v16bf afr[NKT];
#pragma unroll
    for (int kt = 0; kt < NKT; ++kt)
        afr[kt] = load_afrag(&sX[rowA * K + kt * 32 + kbaseA]);

#pragma unroll 4
    for (int nt = 0; nt < 32; ++nt) {
        const int n = nt * 16 + l15;
        v8f acc = {};
#pragma unroll
        for (int kt = 0; kt < NKT; ++kt) {
            v16bf bfr = load_bfrag_f32(&Wih[(size_t)n * K + kt * 32 + kbaseB]);
            acc = wmma_bf16(afr[kt], bfr, acc);
        }
        const float bias = sBias[n];
#pragma unroll
        for (int j = 0; j < 8; ++j) {
            const size_t m = m0 + (size_t)w * 16 + j + mel;
            __builtin_nontemporal_store(acc[j] + bias, &xg[m * 512 + n]);
        }
    }
}

// ---------------------------------------------------------------------------
// Kernel 2: the recurrence. One block per 16-row batch tile (grid = 32).
//   Wave w owns hidden columns [w*16, w*16+16) across all four gates, so its
//   four accumulators (i,f,g,o) are element-aligned for the pointwise math.
//   Whh (bf16 fragments) lives entirely in VGPRs (4 gates x 4 K-tiles x 8
//   regs = 128 VGPRs / wave, loaded once). h ping-pongs through a 4 KB LDS
//   tile; c stays in VGPRs. Per step: 16 v_wmma_f32_16x16x32_bf16 per wave.
//
//   Block = 8 waves = exactly 2 waves/SIMD on the WGP, so request min 2
//   waves/EU -> VGPR budget RF/2 (~ >=512), enough to keep the weight
//   fragments + a double-buffered xg tile resident with NO scratch spills.
//   xg for step t+1 is prefetched while step t's WMMAs run.
// ---------------------------------------------------------------------------
__global__ void
__launch_bounds__(256, 2)
__attribute__((amdgpu_waves_per_eu(2)))
lstm_recur_kernel(const float* __restrict__ xg,
                  const float* __restrict__ Whh,
                  float* __restrict__ out,
                  float* __restrict__ hNl) {
    __shared__ __attribute__((aligned(16))) __bf16 sH[16 * 128];

    const int tid  = threadIdx.x;
    const int lane = tid & 31;
    const int w    = tid >> 5;
    const int b0   = blockIdx.x * 16;

    const int l15    = lane & 15;
    const int kbaseA = (lane < 16) ? 0 : 8;
    const int kbaseB = (lane < 16) ? 0 : 8;
    const int mel    = (lane & 16) ? 8 : 0;
    const int hc     = w * 16;  // this wave's hidden-column chunk

    // Resident recurrent-weight fragments: row n = gate*128 + hc + l15.
    v16bf Bfr[4][4];
#pragma unroll
    for (int g = 0; g < 4; ++g)
#pragma unroll
        for (int kt = 0; kt < 4; ++kt)
            Bfr[g][kt] = load_bfrag_f32(
                &Whh[(size_t)(g * 128 + hc + l15) * 128 + kt * 32 + kbaseB]);

    for (int i = tid; i < 16 * 128; i += 256) sH[i] = (__bf16)0.0f;

    float c[8];
#pragma unroll
    for (int j = 0; j < 8; ++j) c[j] = 0.0f;

    // Per-lane xg base: element (g, j, t) lives at constant offsets
    // j*912*512 + g*128 from this base, advancing 512 floats per step.
    const float* xgb = xg + ((size_t)(b0 + mel) * 912) * 512 + hc + l15;

    // Prefetch step-0 gate pre-activations.
    v8f nxt[4];
#pragma unroll
    for (int g = 0; g < 4; ++g)
#pragma unroll
        for (int j = 0; j < 8; ++j)
            nxt[g][j] = __builtin_nontemporal_load(
                xgb + (size_t)j * 912 * 512 + g * 128);

    __syncthreads();

    for (int t = 0; t < 912; ++t) {
        // C := xg_t chunk (already contains bih+bhh)
        v8f acc[4];
#pragma unroll
        for (int g = 0; g < 4; ++g) acc[g] = nxt[g];

        // Prefetch step t+1 while the WMMAs for step t execute.
        if (t + 1 < 912) {
            const float* p = xgb + (size_t)(t + 1) * 512;
#pragma unroll
            for (int g = 0; g < 4; ++g)
#pragma unroll
                for (int j = 0; j < 8; ++j)
                    nxt[g][j] = __builtin_nontemporal_load(
                        p + (size_t)j * 912 * 512 + g * 128);
        }

        // acc += h @ Whh^T   (K = 128 as 4 bf16 WMMA k-steps, 4 gates each)
#pragma unroll
        for (int kt = 0; kt < 4; ++kt) {
            v16bf a = load_afrag(&sH[l15 * 128 + kt * 32 + kbaseA]);
#pragma unroll
            for (int g = 0; g < 4; ++g)
                acc[g] = wmma_bf16(a, Bfr[g][kt], acc[g]);
        }
        __syncthreads();  // all waves done reading sH for step t

#pragma unroll
        for (int j = 0; j < 8; ++j) {
            const float iv = fast_sigmoid(acc[0][j]);
            const float fv = fast_sigmoid(acc[1][j]);
            const float gv = fast_tanh(acc[2][j]);
            const float ov = fast_sigmoid(acc[3][j]);
            const float cn = fv * c[j] + iv * gv;
            c[j] = cn;
            const float hv = ov * fast_tanh(cn);
            const int ml = j + mel;
            out[((size_t)(b0 + ml) * 912 + t) * 128 + hc + l15] = hv;
            sH[ml * 128 + hc + l15] = (__bf16)hv;
            if (t == 911) hNl[(size_t)(b0 + ml) * 128 + hc + l15] = hv;
        }
        __syncthreads();  // new h visible before step t+1
    }
}

// ---------------------------------------------------------------------------
// Kernel 3: the four 128->4 heads over hN [4,512,128]; doubled activations.
//   Output order: (opt, tp, sl, lot), each [4,512,4].
// ---------------------------------------------------------------------------
static __device__ __forceinline__ void softmax4(const float* z, float* o) {
    float m = fmaxf(fmaxf(z[0], z[1]), fmaxf(z[2], z[3]));
    float e0 = __expf(z[0] - m), e1 = __expf(z[1] - m);
    float e2 = __expf(z[2] - m), e3 = __expf(z[3] - m);
    float s = e0 + e1 + e2 + e3;
    o[0] = e0 / s; o[1] = e1 / s; o[2] = e2 / s; o[3] = e3 / s;
}

__global__ void heads_kernel(const float* __restrict__ hN,
                             const float* __restrict__ Wopt, const float* __restrict__ bopt,
                             const float* __restrict__ Wtp,  const float* __restrict__ btp,
                             const float* __restrict__ Wsl,  const float* __restrict__ bsl,
                             const float* __restrict__ Wlot, const float* __restrict__ blot,
                             float* __restrict__ outp) {
    const int r = blockIdx.x * blockDim.x + threadIdx.x;  // layer*512 + b
    if (r >= 4 * 512) return;
    const float* hrow = hN + (size_t)r * 128;

    float z[4][4];
#pragma unroll
    for (int j = 0; j < 4; ++j) {
        z[0][j] = bopt[j]; z[1][j] = btp[j]; z[2][j] = bsl[j]; z[3][j] = blot[j];
    }
    for (int k = 0; k < 128; ++k) {
        const float hv = hrow[k];
#pragma unroll
        for (int j = 0; j < 4; ++j) {
            z[0][j] = fmaf(hv, Wopt[j * 128 + k], z[0][j]);
            z[1][j] = fmaf(hv, Wtp [j * 128 + k], z[1][j]);
            z[2][j] = fmaf(hv, Wsl [j * 128 + k], z[2][j]);
            z[3][j] = fmaf(hv, Wlot[j * 128 + k], z[3][j]);
        }
    }
    // opt: softmax applied twice
    float s1[4], s2[4];
    softmax4(z[0], s1);
    softmax4(s1, s2);
#pragma unroll
    for (int j = 0; j < 4; ++j) outp[0 * 8192 + r * 4 + j] = s2[j];
    // tp, sl, lot: sigmoid applied twice
#pragma unroll
    for (int h = 1; h < 4; ++h)
#pragma unroll
        for (int j = 0; j < 4; ++j)
            outp[h * 8192 + r * 4 + j] = fast_sigmoid(fast_sigmoid(z[h][j]));
}

// ---------------------------------------------------------------------------
// Launch: layer0 gates+recur, then 3 more layers ping-ponging buffers, heads.
// Workspace layout (floats):
//   xg   : 466944*512 = 239,075,328
//   bufA : 466944*128 =  59,768,832
//   bufB : 466944*128 =  59,768,832
//   hN   : 4*512*128  =     262,144
// ---------------------------------------------------------------------------
extern "C" void kernel_launch(void* const* d_in, const int* in_sizes, int n_in,
                              void* d_out, int out_size, void* d_ws, size_t ws_size,
                              hipStream_t stream) {
    const float* x      = (const float*)d_in[0];
    const float* Wih0   = (const float*)d_in[1];
    const float* Whh0   = (const float*)d_in[2];
    const float* bih0   = (const float*)d_in[3];
    const float* bhh0   = (const float*)d_in[4];
    const float* Wih123 = (const float*)d_in[5];
    const float* Whh123 = (const float*)d_in[6];
    const float* bih123 = (const float*)d_in[7];
    const float* bhh123 = (const float*)d_in[8];
    const float* Wopt = (const float*)d_in[9];  const float* bopt = (const float*)d_in[10];
    const float* Wlot = (const float*)d_in[11]; const float* blot = (const float*)d_in[12];
    const float* Wtp  = (const float*)d_in[13]; const float* btp  = (const float*)d_in[14];
    const float* Wsl  = (const float*)d_in[15]; const float* bsl  = (const float*)d_in[16];

    float* ws   = (float*)d_ws;
    float* xg   = ws;
    float* bufA = xg + (size_t)466944 * 512;
    float* bufB = bufA + (size_t)466944 * 128;
    float* hN   = bufB + (size_t)466944 * 128;

    const dim3 blk(256);
    const dim3 gGates(3648);  // 466944 / 128
    const dim3 gRecur(32);    // 512 / 16

    // Layer 0 (K = 32)
    lstm_gates_kernel<32><<<gGates, blk, 0, stream>>>(x, Wih0, bih0, bhh0, xg);
    lstm_recur_kernel<<<gRecur, blk, 0, stream>>>(xg, Whh0, bufA, hN);

    // Layers 1..3 (K = 128), ping-pong bufA/bufB
    float* cur = bufA;
    float* nxt = bufB;
    for (int l = 0; l < 3; ++l) {
        lstm_gates_kernel<128><<<gGates, blk, 0, stream>>>(
            cur, Wih123 + (size_t)l * 512 * 128, bih123 + l * 512,
            bhh123 + l * 512, xg);
        lstm_recur_kernel<<<gRecur, blk, 0, stream>>>(
            xg, Whh123 + (size_t)l * 512 * 128, nxt,
            hN + (size_t)(l + 1) * 512 * 128);
        float* tmp = cur; cur = nxt; nxt = tmp;
    }

    heads_kernel<<<dim3(8), blk, 0, stream>>>(hN, Wopt, bopt, Wtp, btp,
                                              Wsl, bsl, Wlot, blot,
                                              (float*)d_out);
}
